// LSTMForecast_55533927137578
// MI455X (gfx1250) — compile-verified, matrix operand order checked by reading, not run on hardware
//
#include <hip/hip_runtime.h>

typedef __attribute__((ext_vector_type(16))) _Float16 v16h;
typedef __attribute__((ext_vector_type(8)))  float    v8f;

#define B_TOT 1024
#define SEQ   512
#define HID   50
#define G4    200        // 4*H gate columns
#define NT_LIVE 13       // n-tiles containing real gate columns
#define NROW  208        // gates row stride (floats): 200 live + 8 pad, 16B-aligned rows
#define KPAD  64         // K=50 padded to 64 (two 16x16x32 k-tiles)
#define TPB   512        // 16 wave32 per block; one WMMA n-tile per wave
#define CELLS (16 * HID) // 800 LSTM cells per batch tile

union Frag16 { v16h v; unsigned int u[8]; _Float16 h[16]; };

__device__ __forceinline__ float tanh_f(float x) {
#if __has_builtin(__builtin_amdgcn_tanhf)
    return __builtin_amdgcn_tanhf(x);           // v_tanh_f32 (TRANS pipe)
#else
    return 1.0f - 2.0f / (__expf(2.0f * x) + 1.0f);
#endif
}
__device__ __forceinline__ float sigmoid_f(float x) {
    return __builtin_fmaf(tanh_f(0.5f * x), 0.5f, 0.5f);
}

__global__ __launch_bounds__(TPB)
void lstm_wmma_kernel(const float* __restrict__ x,   const float* __restrict__ h0,
                      const float* __restrict__ c0,  const float* __restrict__ Wih,
                      const float* __restrict__ Whh, const float* __restrict__ bih,
                      const float* __restrict__ bhh, const float* __restrict__ fcw,
                      const float* __restrict__ fcb, float* __restrict__ out)
{
    __shared__ float    x_lds[SEQ * 16];                              // 32 KB, [s][b] transposed
    __shared__ _Float16 h16[16 * KPAD] __attribute__((aligned(16)));  // f16 h, zero-padded K
    __shared__ float    gates[16 * NROW] __attribute__((aligned(16))); // [b][hh][gate] + row pad
    __shared__ float    wih_pk[HID * 4];                              // [hh][gate] packed
    __shared__ float    bsum_pk[HID * 4];
    __shared__ float    h32[16 * HID];                                // final h for fc head only

    const int tid  = threadIdx.x;
    const int lane = tid & 31;
    const int wave = __builtin_amdgcn_readfirstlane((int)threadIdx.x) >> 5;  // scalar
    const int bm   = blockIdx.x * 16;
    const int m    = lane & 15;
    const int half = (lane >> 4) & 1;

    // ---- B fragments (W_hh^T, f16, zero-padded) straight into registers ----
    Frag16 bf0, bf1;
    const int  n      = wave * 16 + m;           // gate column owned by this lane
    const bool ncheck = (n < G4);
    #pragma unroll
    for (int e = 0; e < 16; ++e) {
        const int v = e >> 1, j = e & 1;
        const int k0 = ((v & 3) * 2) + ((v >> 2) * 16) + half * 8 + j;
        const int k1 = k0 + 32;
        bf0.h[e] = (_Float16)((ncheck && k0 < HID) ? Whh[n * HID + k0] : 0.0f);
        bf1.h[e] = (_Float16)((ncheck && k1 < HID) ? Whh[n * HID + k1] : 0.0f);
    }
    // Transposed store offset within a gates row: [hh][gate]; pad cols go to 200..207.
    const int noff = (n < G4) ? ((n % HID) * 4 + (n / HID)) : n;

    // ---- one-time setup ----
    for (int i = tid; i < 16 * SEQ; i += TPB)    // coalesced read, transposed LDS write
        x_lds[(i & (SEQ - 1)) * 16 + (i >> 9)] = x[(size_t)(bm + (i >> 9)) * SEQ + (i & (SEQ - 1))];
    if (tid < G4) {                              // pack W_ih and bias as [hh][gate]
        const int gate = tid / HID, hh = tid % HID;
        wih_pk[hh * 4 + gate]  = Wih[tid];
        bsum_pk[hh * 4 + gate] = bih[tid] + bhh[tid];
    }
    for (int i = tid; i < 16 * KPAD; i += TPB) h16[i] = (_Float16)0.0f;
    __syncthreads();

    // Each thread owns cell A = tid, and (waves 0..8) cell B = tid + 512.
    const int  cA = tid;
    const int  bA = cA / HID, hA = cA % HID;
    const bool hasB = (wave < 9);                // tid < 288, uniform per wave
    const int  cB = tid + TPB;
    const int  bB = (hasB ? cB : 0) / HID, hB = (hasB ? cB : 0) % HID;

    float hregA = h0[(size_t)(bm + bA) * HID + hA];
    float cregA = c0[(size_t)(bm + bA) * HID + hA];
    float hregB = 0.0f, cregB = 0.0f;
    if (hasB) {
        hregB = h0[(size_t)(bm + bB) * HID + hB];
        cregB = c0[(size_t)(bm + bB) * HID + hB];
    }
    h16[bA * KPAD + hA] = (_Float16)hregA;
    if (hasB) h16[bB * KPAD + hB] = (_Float16)hregB;

    const float4 wpA = *(const float4*)&wih_pk[hA * 4];
    const float4 bpA = *(const float4*)&bsum_pk[hA * 4];
    const float4 wpB = *(const float4*)&wih_pk[hB * 4];
    const float4 bpB = *(const float4*)&bsum_pk[hB * 4];
    const int gA = bA * NROW + hA * 4;           // loop-invariant gate addresses
    const int gB = bB * NROW + hB * 4;
    __syncthreads();

    // ---- sequential scan: no global memory, 2 barriers per step ----
    for (int s = 0; s < SEQ; ++s) {
        // A fragments (h tile, 16x64 f16) from LDS
        Frag16 a0, a1;
        #pragma unroll
        for (int v = 0; v < 8; ++v) {
            const int kk = ((v & 3) * 2) + ((v >> 2) * 16) + half * 8;
            a0.u[v] = *(const unsigned int*)&h16[m * KPAD + kk];
            a1.u[v] = *(const unsigned int*)&h16[m * KPAD + 32 + kk];
        }

        if (wave < NT_LIVE) {                    // scalar branch; EXEC all-1s inside
            v8f acc = {};
            acc = __builtin_amdgcn_wmma_f32_16x16x32_f16(false, a0.v, false, bf0.v,
                                                         (short)0, acc, false, false);
            acc = __builtin_amdgcn_wmma_f32_16x16x32_f16(false, a1.v, false, bf1.v,
                                                         (short)0, acc, false, false);
            // D: N = lane%16 -> column noff, M = r + 8*half -> batch row
            #pragma unroll
            for (int r = 0; r < 8; ++r)
                gates[(r + half * 8) * NROW + noff] = acc[r];
        }
        __syncthreads();

        // elementwise cell update: 1x b128 + 1x b32 loads, 1x b16 store per cell
        {
            const float4 g  = *(const float4*)&gates[gA];
            const float  xv = x_lds[s * 16 + bA];
            const float ig = g.x + xv * wpA.x + bpA.x;
            const float fg = g.y + xv * wpA.y + bpA.y;
            const float gg = g.z + xv * wpA.z + bpA.z;
            const float og = g.w + xv * wpA.w + bpA.w;
            const float cn = sigmoid_f(fg) * cregA + sigmoid_f(ig) * tanh_f(gg);
            hregA = sigmoid_f(og) * tanh_f(cn);
            cregA = cn;
            h16[bA * KPAD + hA] = (_Float16)hregA;
        }
        if (hasB) {
            const float4 g  = *(const float4*)&gates[gB];
            const float  xv = x_lds[s * 16 + bB];
            const float ig = g.x + xv * wpB.x + bpB.x;
            const float fg = g.y + xv * wpB.y + bpB.y;
            const float gg = g.z + xv * wpB.z + bpB.z;
            const float og = g.w + xv * wpB.w + bpB.w;
            const float cn = sigmoid_f(fg) * cregB + sigmoid_f(ig) * tanh_f(gg);
            hregB = sigmoid_f(og) * tanh_f(cn);
            cregB = cn;
            h16[bB * KPAD + hB] = (_Float16)hregB;
        }
        __syncthreads();
    }

    // ---- outputs: h_n and c_n straight from registers; fc head via LDS ----
    h32[cA] = hregA;
    out[B_TOT + (size_t)(bm + bA) * HID + hA] = hregA;
    out[B_TOT + (size_t)B_TOT * HID + (size_t)(bm + bA) * HID + hA] = cregA;
    if (hasB) {
        h32[cB] = hregB;
        out[B_TOT + (size_t)(bm + bB) * HID + hB] = hregB;
        out[B_TOT + (size_t)B_TOT * HID + (size_t)(bm + bB) * HID + hB] = cregB;
    }
    __syncthreads();

    if (tid < 16) {
        float acc = fcb[0];
        #pragma unroll 10
        for (int k = 0; k < HID; ++k) acc += h32[tid * HID + k] * fcw[k];
        out[bm + tid] = acc;
    }
}

extern "C" void kernel_launch(void* const* d_in, const int* in_sizes, int n_in,
                              void* d_out, int out_size, void* d_ws, size_t ws_size,
                              hipStream_t stream) {
    const float* x    = (const float*)d_in[0];
    const float* h0   = (const float*)d_in[1];
    const float* c0   = (const float*)d_in[2];
    const float* Wih  = (const float*)d_in[3];
    const float* Whh  = (const float*)d_in[4];
    const float* bih  = (const float*)d_in[5];
    const float* bhh  = (const float*)d_in[6];
    const float* fcw  = (const float*)d_in[7];
    const float* fcb  = (const float*)d_in[8];
    float* out = (float*)d_out;

    dim3 grid(B_TOT / 16);   // 64 batch tiles
    dim3 block(TPB);         // 16 wave32 per tile
    lstm_wmma_kernel<<<grid, block, 0, stream>>>(x, h0, c0, Wih, Whh, bih, bhh,
                                                 fcw, fcb, out);
}
